// DHSFNN_6038724018534
// MI455X (gfx1250) — matrix operand order repaired and checked by reading, not compiled, not run on hardware
//
#include <hip/hip_runtime.h>

// ---------------------------------------------------------------------------
// DH-SNN persistent kernel for MI455X (gfx1250, wave32, WMMA).
// B=256, T=250, IN=700, H=256, BR=4, OUT=20.
// Grid: 64 WGs x 512 threads (16 waves). WG g owns hidden columns
// [16g,16g+16) = neurons [4g,4g+4) x 4 branches. Wave w owns batch rows
// [16w,16w+16). Per step: wmma-GEMM1 (K=704 padded), neuron update,
// grid barrier, wmma-GEMM2 (K=256), neuron update, grid barrier,
// readout GEMM + softmax accumulate (WG 0 only).
// XH=true variant reads x from a pre-converted fp16 [T][B][704] buffer
// (half the A-fragment bytes, no cvt VALU, no tail guard); selected at
// launch time iff the workspace is large enough.
// ---------------------------------------------------------------------------

#define B_   256
#define T_   250
#define IN_  700
#define INP_ 704
#define H_   256
#define OUT_ 20
#define NWG  64
#define NTHR 512
#define KS1  (INP_/32)   // 22 wmma k-steps, layer 1
#define KS2  (H_/32)     // 8  wmma k-steps, layer 2 / readout
#define LD1  712         // LDS row stride (halfs) for W1 slice (bank-friendly)
#define LD2  264         // LDS row stride (halfs) for W2/Wr slices

typedef __attribute__((ext_vector_type(16))) _Float16 v16h;
typedef __attribute__((ext_vector_type(8)))  _Float16 v8h;
typedef __attribute__((ext_vector_type(8)))  float    v8f;

__device__ __forceinline__ float sigm(float x) { return 1.0f / (1.0f + __expf(-x)); }

__device__ __forceinline__ v16h cat8(v8h lo, v8h hi) {
  return __builtin_shufflevector(lo, hi, 0,1,2,3,4,5,6,7,8,9,10,11,12,13,14,15);
}

__device__ __forceinline__ v8f wmma16(v16h a, v16h b, v8f c) {
  return __builtin_amdgcn_wmma_f32_16x16x32_f16(false, a, false, b, (short)0, c, false, false);
}

// Sense-reversal device-wide barrier. cnt/gen live in workspace, memset to 0
// at the start of every kernel_launch (graph-capturable), so replays are
// deterministic: cnt returns to 0 at every completion, gen just increments.
__device__ __forceinline__ void grid_barrier(unsigned* cnt, unsigned* gen) {
  __syncthreads();
  if (threadIdx.x == 0) {
    __threadfence();
    unsigned g = __hip_atomic_load(gen, __ATOMIC_RELAXED, __HIP_MEMORY_SCOPE_AGENT);
    unsigned a = __hip_atomic_fetch_add(cnt, 1u, __ATOMIC_ACQ_REL, __HIP_MEMORY_SCOPE_AGENT) + 1u;
    if (a == (unsigned)NWG) {
      __hip_atomic_store(cnt, 0u, __ATOMIC_RELAXED, __HIP_MEMORY_SCOPE_AGENT);
      __hip_atomic_store(gen, g + 1u, __ATOMIC_RELEASE, __HIP_MEMORY_SCOPE_AGENT);
    } else {
      while (__hip_atomic_load(gen, __ATOMIC_ACQUIRE, __HIP_MEMORY_SCOPE_AGENT) == g) {
        __builtin_amdgcn_s_sleep(1);
      }
    }
    __threadfence();
  }
  __syncthreads();
}

// x fp32 [B][T][IN] -> fp16 [T][B][INP_] (K padded with zeros)
__global__ __launch_bounds__(256)
void convert_x(const float* __restrict__ x, _Float16* __restrict__ x16) {
  size_t i = (size_t)blockIdx.x * 256 + threadIdx.x;
  const size_t n = (size_t)T_ * B_ * INP_;
  if (i >= n) return;
  int k = (int)(i % INP_);
  size_t tb = i / INP_;
  int b = (int)(tb % B_);
  int t = (int)(tb / B_);
  float v = (k < IN_) ? x[((size_t)b * T_ + t) * IN_ + k] : 0.0f;
  x16[i] = (_Float16)v;
}

template<bool XH>
__global__ __launch_bounds__(NTHR)
void dhsnn_persistent(const float* __restrict__ x,
                      const _Float16* __restrict__ x16,
                      const float* __restrict__ W1, const float* __restrict__ b1,
                      const float* __restrict__ tn1, const float* __restrict__ tm1,
                      const float* __restrict__ W2, const float* __restrict__ b2,
                      const float* __restrict__ tn2, const float* __restrict__ tm2,
                      const float* __restrict__ Wr, const float* __restrict__ br,
                      const float* __restrict__ tmr,
                      float* __restrict__ out,
                      _Float16* __restrict__ s1buf,   // [B_, H_] f16
                      _Float16* __restrict__ s2buf,   // [B_, H_] f16
                      unsigned* __restrict__ sync)
{
  __shared__ _Float16 W1lds[16 * LD1];   // W1 rows [16g,16g+16), K padded to 704
  __shared__ _Float16 W2lds[16 * LD2];   // W2 rows [16g,16g+16), K=256
  __shared__ _Float16 Wrlds[32 * LD2];   // Wr rows 0..31 (>=20 zero), WG0 only

  const int g    = blockIdx.x;
  const int tid  = threadIdx.x;
  const int lane = tid & 31;
  const int w    = tid >> 5;
  const int l15  = lane & 15;
  const int hi   = lane >> 4;
  const int m0   = w * 16;
  const int kb   = hi * 8;               // A-fragment K base offset per lane half

  // ---- stage weights into LDS (fp32 -> fp16) -------------------------------
  for (int i = tid; i < 16 * INP_; i += NTHR) {
    int r = i / INP_, k = i - r * INP_;
    float v = (k < IN_) ? W1[(size_t)(16 * g + r) * IN_ + k] : 0.0f;
    W1lds[r * LD1 + k] = (_Float16)v;
  }
  for (int i = tid; i < 16 * H_; i += NTHR) {
    int r = i >> 8, k = i & 255;
    W2lds[r * LD2 + k] = (_Float16)W2[(size_t)(16 * g + r) * H_ + k];
  }
  if (g == 0) {
    for (int i = tid; i < 32 * H_; i += NTHR) {
      int r = i >> 8, k = i & 255;
      Wrlds[r * LD2 + k] = (r < OUT_) ? (_Float16)Wr[(size_t)r * H_ + k] : (_Float16)0.0f;
    }
  }
  __syncthreads();

  // ---- per-lane constants ---------------------------------------------------
  const int n_glob = 16 * g + l15;             // hidden column (h*4 + branch)
  const int h_glob = 4 * g + (l15 >> 2);       // neuron index
  const float b1v = b1[n_glob];
  const float be1 = sigm(tn1[n_glob]);
  const float a1v = sigm(tm1[h_glob]);
  const float b2v = b2[n_glob];
  const float be2 = sigm(tn2[n_glob]);
  const float a2v = sigm(tm2[h_glob]);
  float br0 = 0.f, ar0 = 0.f, br1 = 0.f, ar1 = 0.f;
  if (g == 0) {
    br0 = br[l15]; ar0 = sigm(tmr[l15]);       // cols 0..15 always valid
    if (16 + l15 < OUT_) { br1 = br[16 + l15]; ar1 = sigm(tmr[16 + l15]); }
  }
  const bool val1 = (l15 < 4);

  // ---- recurrent state in registers (WMMA C/D layout) -----------------------
  float d1[8] = {}, m1[8] = {}, s1[8] = {};
  float d2[8] = {}, m2[8] = {}, s2[8] = {};
  float mr0[8] = {}, mr1[8] = {}, ao0[8] = {}, ao1[8] = {};

  unsigned* cnt = sync;
  unsigned* gen = sync + 16;

  const float* xrowf = x + ((size_t)(m0 + l15) * T_) * IN_;

  for (int t = 0; t < T_; ++t) {
    // ================= phase 1: hidden layer 1 ==============================
    const float*    xr   = xrowf + (size_t)t * IN_;
    const _Float16* xr16 = x16 + ((size_t)t * B_ + (m0 + l15)) * INP_;

    auto loadA1 = [&](int ks) -> v16h {
      const int k0 = ks * 32 + kb;
      if constexpr (XH) {
        return cat8(*(const v8h*)(xr16 + k0), *(const v8h*)(xr16 + k0 + 16));
      } else {
        v16h a;
        if (ks < KS1 - 1) {
          float4 f0 = *(const float4*)(xr + k0);
          float4 f1 = *(const float4*)(xr + k0 + 4);
          float4 f2 = *(const float4*)(xr + k0 + 16);
          float4 f3 = *(const float4*)(xr + k0 + 20);
          a[0]=(_Float16)f0.x;  a[1]=(_Float16)f0.y;  a[2]=(_Float16)f0.z;  a[3]=(_Float16)f0.w;
          a[4]=(_Float16)f1.x;  a[5]=(_Float16)f1.y;  a[6]=(_Float16)f1.z;  a[7]=(_Float16)f1.w;
          a[8]=(_Float16)f2.x;  a[9]=(_Float16)f2.y;  a[10]=(_Float16)f2.z; a[11]=(_Float16)f2.w;
          a[12]=(_Float16)f3.x; a[13]=(_Float16)f3.y; a[14]=(_Float16)f3.z; a[15]=(_Float16)f3.w;
        } else {
          #pragma unroll
          for (int j = 0; j < 8; ++j) {     // guard the 700..703 pad region
            int ka = k0 + j, kc = k0 + 16 + j;
            a[j]     = (_Float16)((ka < IN_) ? xr[ka] : 0.0f);
            a[8 + j] = (_Float16)((kc < IN_) ? xr[kc] : 0.0f);
          }
        }
        return a;
      }
    };
    auto loadB1 = [&](int ks) -> v16h {
      const _Float16* bp = &W1lds[l15 * LD1 + ks * 32 + hi * 8];
      return cat8(*(const v8h*)bp, *(const v8h*)(bp + 16));
    };

    // software-pipelined, dual accumulator chains (breaks RAW/WAR hazards)
    v8f acc0 = {}, acc1 = {};
    {
      v16h aC = loadA1(0), bC = loadB1(0);
      #pragma unroll
      for (int ks = 0; ks < KS1; ++ks) {
        v16h aN, bN;
        if (ks + 1 < KS1) { aN = loadA1(ks + 1); bN = loadB1(ks + 1); }
        if (ks & 1) acc1 = wmma16(aC, bC, acc1);
        else        acc0 = wmma16(aC, bC, acc0);
        if (ks + 1 < KS1) { aC = aN; bC = bN; }
      }
    }
    #pragma unroll
    for (int v = 0; v < 8; ++v) {
      float raw = (acc0[v] + acc1[v]) + b1v;
      d1[v] = be1 * d1[v] + (1.0f - be1) * raw;
      float s = d1[v] + __shfl_xor(d1[v], 1, 32);   // branch sum over 4 lanes
      s += __shfl_xor(s, 2, 32);
      m1[v] = (m1[v] - s1[v]) * a1v + (1.0f - a1v) * s;   // VTH = 1
      s1[v] = (m1[v] > 1.0f) ? 1.0f : 0.0f;
    }
    if ((lane & 3) == 0) {
      #pragma unroll
      for (int v = 0; v < 8; ++v)
        s1buf[(size_t)(m0 + v + 8 * hi) * H_ + h_glob] = (_Float16)s1[v];
    }
    grid_barrier(cnt, gen);

    // ================= phase 2: hidden layer 2 ==============================
    const _Float16* s1r = s1buf + (size_t)(m0 + l15) * H_;
    auto loadA2 = [&](int ks) -> v16h {
      const int k0 = ks * 32 + kb;
      return cat8(*(const v8h*)(s1r + k0), *(const v8h*)(s1r + k0 + 16));
    };
    auto loadB2 = [&](int ks) -> v16h {
      const _Float16* bp = &W2lds[l15 * LD2 + ks * 32 + hi * 8];
      return cat8(*(const v8h*)bp, *(const v8h*)(bp + 16));
    };
    v8f bcc0 = {}, bcc1 = {};
    {
      v16h aC = loadA2(0), bC = loadB2(0);
      #pragma unroll
      for (int ks = 0; ks < KS2; ++ks) {
        v16h aN, bN;
        if (ks + 1 < KS2) { aN = loadA2(ks + 1); bN = loadB2(ks + 1); }
        if (ks & 1) bcc1 = wmma16(aC, bC, bcc1);
        else        bcc0 = wmma16(aC, bC, bcc0);
        if (ks + 1 < KS2) { aC = aN; bC = bN; }
      }
    }
    #pragma unroll
    for (int v = 0; v < 8; ++v) {
      float raw = (bcc0[v] + bcc1[v]) + b2v;
      d2[v] = be2 * d2[v] + (1.0f - be2) * raw;
      float s = d2[v] + __shfl_xor(d2[v], 1, 32);
      s += __shfl_xor(s, 2, 32);
      m2[v] = (m2[v] - s2[v]) * a2v + (1.0f - a2v) * s;
      s2[v] = (m2[v] > 1.0f) ? 1.0f : 0.0f;
    }
    if ((lane & 3) == 0) {
      #pragma unroll
      for (int v = 0; v < 8; ++v)
        s2buf[(size_t)(m0 + v + 8 * hi) * H_ + h_glob] = (_Float16)s2[v];
    }
    grid_barrier(cnt, gen);

    // ============ phase 3: readout + softmax accumulate (WG 0 only) =========
    // s2buf is next written only after the next grid barrier, which WG0 must
    // also reach, so this overlaps safely with other WGs' next phase 1.
    if (g == 0) {
      const _Float16* s2r = s2buf + (size_t)(m0 + l15) * H_;
      v8f r0 = {}, r1 = {};                 // two independent chains already
      #pragma unroll
      for (int ks = 0; ks < KS2; ++ks) {
        const int k0 = ks * 32 + kb;
        v16h a = cat8(*(const v8h*)(s2r + k0), *(const v8h*)(s2r + k0 + 16));
        const _Float16* bp0 = &Wrlds[l15 * LD2 + ks * 32 + hi * 8];
        const _Float16* bp1 = &Wrlds[(16 + l15) * LD2 + ks * 32 + hi * 8];
        v16h bf0 = cat8(*(const v8h*)bp0, *(const v8h*)(bp0 + 16));
        v16h bf1 = cat8(*(const v8h*)bp1, *(const v8h*)(bp1 + 16));
        r0 = wmma16(a, bf0, r0);
        r1 = wmma16(a, bf1, r1);
      }
      const float gate = (t > 3) ? 1.0f : 0.0f;   // WARMUP = 3
      #pragma unroll
      for (int v = 0; v < 8; ++v) {
        mr0[v] = mr0[v] * ar0 + (1.0f - ar0) * (r0[v] + br0);
        mr1[v] = mr1[v] * ar1 + (1.0f - ar1) * (r1[v] + br1);
        float v0 = mr0[v];
        float v1 = val1 ? mr1[v] : -1e30f;
        float mx = fmaxf(v0, v1);
        mx = fmaxf(mx, __shfl_xor(mx, 1, 32));
        mx = fmaxf(mx, __shfl_xor(mx, 2, 32));
        mx = fmaxf(mx, __shfl_xor(mx, 4, 32));
        mx = fmaxf(mx, __shfl_xor(mx, 8, 32));
        float e0 = __expf(v0 - mx);
        float e1 = val1 ? __expf(v1 - mx) : 0.0f;
        float sm = e0 + e1;
        sm += __shfl_xor(sm, 1, 32);
        sm += __shfl_xor(sm, 2, 32);
        sm += __shfl_xor(sm, 4, 32);
        sm += __shfl_xor(sm, 8, 32);
        float inv = 1.0f / sm;
        ao0[v] += gate * e0 * inv;
        ao1[v] += gate * e1 * inv;
      }
    }
  }

  // ---- write accumulated softmax readout [B_, OUT_] -------------------------
  if (g == 0) {
    #pragma unroll
    for (int v = 0; v < 8; ++v) {
      int row = m0 + v + 8 * hi;
      out[row * OUT_ + l15] = ao0[v];
      if (val1) out[row * OUT_ + 16 + l15] = ao1[v];
    }
  }
}

extern "C" void kernel_launch(void* const* d_in, const int* in_sizes, int n_in,
                              void* d_out, int out_size, void* d_ws, size_t ws_size,
                              hipStream_t stream) {
  (void)in_sizes; (void)n_in; (void)out_size;
  const float* x   = (const float*)d_in[0];
  const float* W1  = (const float*)d_in[1];
  const float* b1  = (const float*)d_in[2];
  const float* tn1 = (const float*)d_in[3];
  const float* tm1 = (const float*)d_in[4];
  const float* W2  = (const float*)d_in[5];
  const float* b2  = (const float*)d_in[6];
  const float* tn2 = (const float*)d_in[7];
  const float* tm2 = (const float*)d_in[8];
  const float* Wr  = (const float*)d_in[9];
  const float* brr = (const float*)d_in[10];
  const float* tmr = (const float*)d_in[11];
  float* out = (float*)d_out;

  char* ws = (char*)d_ws;
  unsigned* sync   = (unsigned*)ws;                       // 256 B barrier area
  _Float16* s1buf  = (_Float16*)(ws + 256);               // 256*256*2 B
  _Float16* s2buf  = s1buf + (size_t)B_ * H_;             // 256*256*2 B
  _Float16* x16    = s2buf + (size_t)B_ * H_;             // optional 90.1 MB

  const size_t base_need = 256 + 2 * (size_t)B_ * H_ * sizeof(_Float16);
  const size_t x16_elems = (size_t)T_ * B_ * INP_;
  const bool use_xh = ws_size >= base_need + x16_elems * sizeof(_Float16);

  // reset barrier counters every launch (graph-capturable, deterministic)
  hipMemsetAsync(sync, 0, 256, stream);

  if (use_xh) {
    const int cvt_blocks = (int)((x16_elems + 255) / 256);
    hipLaunchKernelGGL(convert_x, dim3(cvt_blocks), dim3(256), 0, stream, x, x16);
    hipLaunchKernelGGL((dhsnn_persistent<true>), dim3(NWG), dim3(NTHR), 0, stream,
                       x, x16, W1, b1, tn1, tm1, W2, b2, tn2, tm2, Wr, brr, tmr,
                       out, s1buf, s2buf, sync);
  } else {
    hipLaunchKernelGGL((dhsnn_persistent<false>), dim3(NWG), dim3(NTHR), 0, stream,
                       x, (const _Float16*)nullptr, W1, b1, tn1, tm1,
                       W2, b2, tn2, tm2, Wr, brr, tmr,
                       out, s1buf, s2buf, sync);
  }
}
